// SimpleRNN_16252156248547
// MI455X (gfx1250) — compile-verified, matrix operand order checked by reading, not run on hardware
//
#include <hip/hip_runtime.h>
#include <hip/hip_bf16.h>

// ---------------------------------------------------------------------------
// Adaptive ReLU RNN for MI455X (gfx1250, wave32, WMMA + TDM).
//   Phase 0: pack W_ih / W_hh into native WMMA B-operand layout (bf16).
//   Phase 1: x_proj = x @ W_ih^T + b_ih + b_hh  (bf16 WMMA GEMM, fp32 accum,
//            32-row M-tiles, software-pipelined k-loop)
//   Phase 2: persistent 8-WG scan; W_hh slice resident in LDS, h staged each
//            step via TENSOR_LOAD_TO_LDS (TDM, padded 2D tile), adaptation
//            state in VGPRs, device-scope atomic barrier per timestep.
// ---------------------------------------------------------------------------

typedef __bf16 bf16_t;
typedef __attribute__((ext_vector_type(8)))  float        v8f;
typedef __attribute__((ext_vector_type(16))) __bf16       v16bf;
typedef __attribute__((ext_vector_type(8)))  __bf16       v8bf;
typedef __attribute__((ext_vector_type(4)))  __bf16       v4bf;
typedef __attribute__((ext_vector_type(4)))  unsigned int v4u;
typedef __attribute__((ext_vector_type(8)))  int          v8i_t;
typedef __attribute__((ext_vector_type(4)))  int          v4i_t;

#define T_SEQ   2048
#define BATCH   32
#define ISZ     512
#define HSZ     512
#define ADAPT_R 0.2f
#define RECOV_R 0.1f

// LDS pitch for a 512-element bf16 row: 1024B data + 16B pad -> 1040.
// 1040/4 = 260 dwords; 260 % 64 = 4 -> 16 consecutive rows cover all 64
// banks with conflict-free b128 accesses.
#define APITCH 1040

#if defined(__has_builtin)
#if __has_builtin(__builtin_amdgcn_tensor_load_to_lds)
#define HAVE_TDM 1
#endif
#endif

#if defined(__HIP_DEVICE_COMPILE__) && !defined(HAVE_TDM)
#warning "CDNA5 TDM builtin NOT available in device pass - manual LDS staging used"
#endif

static __device__ __forceinline__ v16bf cat16(v8bf lo, v8bf hi) {
  return __builtin_shufflevector(lo, hi, 0,1,2,3,4,5,6,7,8,9,10,11,12,13,14,15);
}

// ---------------------------------------------------------------------------
// Pack a [N=512][K=512] row-major fp32 weight (used as B(k,n) = w[n][k]) into
// the wave32 16-bit B-operand layout: flat[((ntile*16+ktile)*32+lane)*16+e];
// lane holds column n = ntile*16 + (lane&15); k-half kb = (lane>>4)*8;
// element e -> k = ktile*32 + kb + e + (e>=8 ? 8 : 0).
// ---------------------------------------------------------------------------
__global__ __launch_bounds__(256) void pack_b_kernel(const float* __restrict__ w,
                                                     bf16_t* __restrict__ out) {
  int id   = blockIdx.x * 256 + threadIdx.x;   // 0 .. 262143
  int e    = id & 15;
  int lane = (id >> 4) & 31;
  int kt   = (id >> 9) & 15;
  int ntg  = id >> 13;                         // 0..31
  int n    = ntg * 16 + (lane & 15);
  int kb   = (lane >> 4) * 8;
  int k    = kt * 32 + kb + e + ((e >= 8) ? 8 : 0);
  out[id]  = (bf16_t)w[n * 512 + k];
}

// ---------------------------------------------------------------------------
// Phase 1: x_proj[r][h] = sum_i x[r][i]*W_ih[h][i] + b_ih[h] + b_hh[h]
// One block = 32 rows (2 m-tiles); 8 waves each own a 64-column slice and
// reuse each B tile across both m-tiles. k-loop software-pipelined.
// ---------------------------------------------------------------------------
__global__ __launch_bounds__(256) void xproj_gemm_kernel(
    const float*  __restrict__ x,
    const bf16_t* __restrict__ wih_p,
    const float*  __restrict__ bih,
    const float*  __restrict__ bhh,
    float*        __restrict__ xp) {
  __shared__ __align__(16) char lds[32 * APITCH];
  const int tid     = threadIdx.x;
  const int rowbase = blockIdx.x * 32;

  // Stage 32x512 fp32 -> bf16 into padded LDS (4096 float4 chunks, coalesced).
  #pragma unroll
  for (int p = 0; p < 16; ++p) {
    int c    = p * 256 + tid;
    int row  = c >> 7;           // 128 float4 per row
    int col4 = c & 127;
    const float4 f = ((const float4*)(x + (size_t)(rowbase + row) * ISZ))[col4];
    v4bf b;
    b[0] = (bf16_t)f.x; b[1] = (bf16_t)f.y; b[2] = (bf16_t)f.z; b[3] = (bf16_t)f.w;
    *(v4bf*)(lds + row * APITCH + col4 * 8) = b;
  }
  __syncthreads();

  const int lane = tid & 31;
  const int wv   = tid >> 5;           // wave32: 8 waves
  const int kb   = (lane >> 4) * 8;

  // A operand base addresses for the two m-tiles (per-kt stride = 64B).
  const char* a0 = lds + (lane & 15) * APITCH + kb * 2;
  const char* a1 = a0 + 16 * APITCH;
  // Packed B: element offset = ntg*8192 + kt*512 + lane*16.
  const bf16_t* bp = wih_p + ((size_t)(wv * 4) * 8192 + (size_t)lane * 16);

  v8f acc[2][4] = {};

  // Software pipeline: prologue loads kt=0 operands.
  v8bf alo0 = *(const v8bf*)a0,        ahi0 = *(const v8bf*)(a0 + 32);
  v8bf alo1 = *(const v8bf*)a1,        ahi1 = *(const v8bf*)(a1 + 32);
  v16bf bt[4];
  #pragma unroll
  for (int nt = 0; nt < 4; ++nt) bt[nt] = *(const v16bf*)(bp + nt * 8192);

  #pragma unroll 4
  for (int kt = 0; kt < 16; ++kt) {
    v16bf A0 = cat16(alo0, ahi0);
    v16bf A1 = cat16(alo1, ahi1);
    v16bf bc[4] = { bt[0], bt[1], bt[2], bt[3] };
    if (kt < 15) {   // issue next iteration's loads before consuming this one's
      alo0 = *(const v8bf*)(a0 + (kt + 1) * 64);
      ahi0 = *(const v8bf*)(a0 + (kt + 1) * 64 + 32);
      alo1 = *(const v8bf*)(a1 + (kt + 1) * 64);
      ahi1 = *(const v8bf*)(a1 + (kt + 1) * 64 + 32);
      #pragma unroll
      for (int nt = 0; nt < 4; ++nt)
        bt[nt] = *(const v16bf*)(bp + nt * 8192 + (kt + 1) * 512);
    }
    #pragma unroll
    for (int nt = 0; nt < 4; ++nt)
      acc[0][nt] = __builtin_amdgcn_wmma_f32_16x16x32_bf16(
          false, A0, false, bc[nt], (short)0, acc[0][nt], false, false);
    #pragma unroll
    for (int nt = 0; nt < 4; ++nt)
      acc[1][nt] = __builtin_amdgcn_wmma_f32_16x16x32_bf16(
          false, A1, false, bc[nt], (short)0, acc[1][nt], false, false);
  }

  // C/D layout: VGPR r -> row M = r + 8*(lane>>4), col N = lane&15.
  #pragma unroll
  for (int nt = 0; nt < 4; ++nt) {
    int n    = wv * 64 + nt * 16 + (lane & 15);
    float bs = bih[n] + bhh[n];
    #pragma unroll
    for (int mi = 0; mi < 2; ++mi) {
      #pragma unroll
      for (int r = 0; r < 8; ++r) {
        int g = rowbase + mi * 16 + ((lane >> 4) << 3) + r;
        xp[(size_t)g * HSZ + n] = acc[mi][nt][r] + bs;
      }
    }
  }
}

// ---------------------------------------------------------------------------
// Phase 2: persistent scan. 8 WGs x 256 threads; block owns 64 hidden cols.
// ---------------------------------------------------------------------------
__global__ __launch_bounds__(256) void rnn_scan_kernel(
    const float*  __restrict__ xp,
    const bf16_t* __restrict__ whh_p,
    float*        __restrict__ out,
    bf16_t*       __restrict__ h_exch,
    unsigned*     __restrict__ syncc) {
  extern __shared__ char smem[];
  char* lds_w = smem;            // 65536 B: packed W_hh B-operand slice
  char* lds_h = smem + 65536;    // 32 * APITCH: staged h (A operand source)

  const int tid    = threadIdx.x;
  const int lane   = tid & 31;
  const int wv     = tid >> 5;
  const int m_tile = wv & 1;     // batch rows [m_tile*16, +16)
  const int nt     = wv >> 1;    // local n-tile 0..3
  const int nb4    = blockIdx.x * 4;

  // One-time: copy this block's contiguous 64KB packed W_hh slice into LDS.
  {
    const uint4* src = (const uint4*)(whh_p + (size_t)nb4 * 16 * 32 * 16);
    uint4* dst = (uint4*)lds_w;
    #pragma unroll
    for (int p = 0; p < 16; ++p) dst[p * 256 + tid] = src[p * 256 + tid];
  }

  const int kb    = (lane >> 4) * 8;
  const int arow  = m_tile * 16 + (lane & 15);
  const int n     = nb4 * 16 + nt * 16 + (lane & 15);
  const int bbase = m_tile * 16 + ((lane >> 4) << 3);
  const unsigned nwg = gridDim.x;
  const char* abase = lds_h + arow * APITCH + kb * 2;            // +kt*64
  const char* bbw   = lds_w + nt * 16384 + lane * 32;            // +kt*1024

#if HAVE_TDM
  // D# group1 (CDNA5 ISA Sec 8.4): data_size=2B (bit16),
  // pad_enable (bit20), pad_interval=7 -> 256 dwords = 1024B (bits24:22),
  // pad_amount=3 -> 4 dwords = 16B (bits31:25). tensor_dim0=512 (bits79:48),
  // tensor_dim1=32 (bits111:80), tile_dim0=512 (bits127:112),
  // tile_dim1=32 (bits143:128), tensor_dim0_stride=512 (bits207:160).
  const v8i_t g1 = { (int)0x07D10000, (int)0x02000000, (int)0x00200000,
                     (int)0x02000000, 32, 512, 0, 0 };
  const v4i_t gz = { 0, 0, 0, 0 };
#if __clang_major__ >= 23
  const v8i_t gz8 = { 0, 0, 0, 0, 0, 0, 0, 0 };
#endif
  const unsigned ldsHBase = (unsigned)(uintptr_t)lds_h;
#endif

  v8f adapt = {};   // per-(b,h) adaptation state; lives in VGPRs all 2048 steps

  for (int t = 0; t < T_SEQ; ++t) {
    const bf16_t* hr = h_exch + (size_t)(t & 1) * (BATCH * HSZ);
    bf16_t*       hw = h_exch + (size_t)((t + 1) & 1) * (BATCH * HSZ);

#if HAVE_TDM
    // Stage h[t] (32x512 bf16, padded rows) via the Tensor Data Mover.
    if (wv == 0) {
      unsigned long long ga = (unsigned long long)(uintptr_t)hr;
      v4u g0;
      g0[0] = 1u;                                   // count=1, user descriptor
      g0[1] = ldsHBase;                             // lds_addr
      g0[2] = (unsigned)ga;                         // global_addr[31:0]
      g0[3] = (unsigned)(ga >> 32) | (2u << 30);    // addr[56:32] | type=2
#if __clang_major__ >= 23
      __builtin_amdgcn_tensor_load_to_lds(g0, g1, gz, gz, gz8, 0);
#else
      __builtin_amdgcn_tensor_load_to_lds(g0, g1, gz, gz, 0);
#endif
      __builtin_amdgcn_s_wait_tensorcnt((short)0);
    }
#else
    {
      const uint4* src = (const uint4*)hr;
      #pragma unroll
      for (int p = 0; p < 8; ++p) {
        int c   = p * 256 + tid;   // 2048 x 16B
        int row = c >> 6;
        int col = c & 63;
        *(uint4*)(lds_h + row * APITCH + col * 16) = src[c];
      }
    }
#endif
    __syncthreads();

    // Prefetch next step's x_proj tile while this step's GEMM runs.
    if (t + 1 < T_SEQ) {
      #pragma unroll
      for (int r = 0; r < 8; r += 2)
        __builtin_prefetch(&xp[((size_t)(t + 1) * BATCH + (bbase + r)) * HSZ + n], 0, 1);
    }

    // Accumulator seeded with x_proj[t] tile (pre = x_proj + h@Whh^T).
    v8f acc;
    #pragma unroll
    for (int r = 0; r < 8; ++r)
      acc[r] = xp[((size_t)t * BATCH + (bbase + r)) * HSZ + n];

    // Software-pipelined K=512 WMMA chain fed from LDS.
    v8bf alo = *(const v8bf*)abase;
    v8bf ahi = *(const v8bf*)(abase + 32);
    v16bf bcur = *(const v16bf*)bbw;
    #pragma unroll 4
    for (int kt = 0; kt < 16; ++kt) {
      v16bf a  = cat16(alo, ahi);
      v16bf bb = bcur;
      if (kt < 15) {
        alo  = *(const v8bf*)(abase + (kt + 1) * 64);
        ahi  = *(const v8bf*)(abase + (kt + 1) * 64 + 32);
        bcur = *(const v16bf*)(bbw + (kt + 1) * 1024);
      }
      acc = __builtin_amdgcn_wmma_f32_16x16x32_bf16(
          false, a, false, bb, (short)0, acc, false, false);
    }

    // Adaptive-threshold ReLU + state update; emit outputs.
    float* outt = out + (size_t)t * BATCH * HSZ;
    #pragma unroll
    for (int r = 0; r < 8; ++r) {
      int   b   = bbase + r;
      float pre = acc[r];
      float ad  = adapt[r];
      float adp = fmaxf(0.0f, pre - ad);
      adapt[r]  = (ad + ADAPT_R * adp) * (1.0f - RECOV_R);
      outt[(size_t)b * HSZ + n] = adp;                        // output[t]
      hw[b * HSZ + n] = (bf16_t)adp;                          // next-step h
      if (t == T_SEQ - 1)
        out[(size_t)T_SEQ * BATCH * HSZ + b * HSZ + n] = adp; // h_last
    }

    // Device-scope step barrier (release our stores, acquire everyone's).
    __threadfence();
    __syncthreads();
    if (tid == 0) {
      __hip_atomic_fetch_add(syncc, 1u, __ATOMIC_ACQ_REL, __HIP_MEMORY_SCOPE_AGENT);
      unsigned target = (unsigned)(t + 1) * nwg;
      while (__hip_atomic_load(syncc, __ATOMIC_ACQUIRE, __HIP_MEMORY_SCOPE_AGENT) < target)
        __builtin_amdgcn_s_sleep(2);
    }
    __syncthreads();
  }
}

// ---------------------------------------------------------------------------
// Workspace layout (bytes):
//   [0,     256)        : sync counter (zeroed each launch)
//   [256,   256+64K)    : h exchange, double-buffered bf16 (zeroed: h0 = 0)
//   next 512K           : packed W_ih (bf16)
//   next 512K           : packed W_hh (bf16)
//   next 128M           : x_proj (fp32)
// ---------------------------------------------------------------------------
extern "C" void kernel_launch(void* const* d_in, const int* in_sizes, int n_in,
                              void* d_out, int out_size, void* d_ws, size_t ws_size,
                              hipStream_t stream) {
  (void)in_sizes; (void)n_in; (void)out_size; (void)ws_size;
  const float* x    = (const float*)d_in[0];
  const float* w_ih = (const float*)d_in[1];
  const float* w_hh = (const float*)d_in[2];
  const float* b_ih = (const float*)d_in[3];
  const float* b_hh = (const float*)d_in[4];
  float* out = (float*)d_out;

  char* ws = (char*)d_ws;
  unsigned* syncc  = (unsigned*)ws;
  bf16_t*   h_exch = (bf16_t*)(ws + 256);
  bf16_t*   wih_p  = (bf16_t*)(ws + 256 + 65536);
  bf16_t*   whh_p  = (bf16_t*)(ws + 256 + 65536 + 524288);
  float*    x_proj = (float*) (ws + 256 + 65536 + 2 * 524288);

  // Zero sync counter + both h buffers (h0 = 0). Graph-capturable.
  hipMemsetAsync(ws, 0, 256 + 65536, stream);

  // Pack weights into WMMA B-operand layout.
  pack_b_kernel<<<1024, 256, 0, stream>>>(w_ih, wih_p);
  pack_b_kernel<<<1024, 256, 0, stream>>>(w_hh, whh_p);

  // Phase 1: input-projection GEMM (2048 row-tiles of 32).
  xproj_gemm_kernel<<<2048, 256, 0, stream>>>(x, wih_p, b_ih, b_hh, x_proj);

  // Phase 2: persistent scan, 8 WGs (one 64-column slice each).
  size_t smem_bytes = 65536 + 32 * APITCH;   // 98816 B
  rnn_scan_kernel<<<8, 256, smem_bytes, stream>>>(x_proj, whh_p, out, h_exch, syncc);
}
